// ESNModule_30734785970265
// MI455X (gfx1250) — matrix operand order repaired and checked by reading, not compiled, not run on hardware
//
#include <hip/hip_runtime.h>
#include <hip/hip_bf16.h>
#include <stdint.h>

// Problem constants (from reference)
#define B_    32
#define T_    1024
#define F_    128
#define H_    2048
#define KAUG  2176          // H_ + F_ : augmented K (state | input)
#define NBLK  64            // H_ / 32 column-tile workgroups for the scan
#define SCAN_THREADS 128    // 4 waves of 32
#define LDSPITCH 2184       // KAUG + 8 : pad so 16-lane row stride is conflict-free

typedef __attribute__((ext_vector_type(16))) __bf16 v16bf;
typedef __attribute__((ext_vector_type(8)))  float  v8f;

union ABPack { v16bf v; uint4 q[2]; };

// round-to-nearest-even f32 -> bf16 bit pattern
__device__ __forceinline__ unsigned short f2bf(float x) {
  unsigned int u = __float_as_uint(x);
  if ((u & 0x7fffffffu) > 0x7f800000u)              // NaN -> quiet NaN
    return (unsigned short)((u >> 16) | 0x0040u);
  u += 0x7fffu + ((u >> 16) & 1u);
  return (unsigned short)(u >> 16);
}

// branch-free tanh: copysign(1 - 2/(exp(2|x|)+1), x)
__device__ __forceinline__ float fast_tanh(float x) {
  const float TWO_LOG2E = 2.885390081777927f;       // 2*log2(e)
  float ax = __builtin_fabsf(x);
  float e  = __builtin_amdgcn_exp2f(ax * TWO_LOG2E);
  float r  = 1.0f - 2.0f * __builtin_amdgcn_rcpf(e + 1.0f);
  return __builtin_copysignf(r, x);
}

// pack two 16B chunks into one WMMA operand
__device__ __forceinline__ v16bf ldpack(const unsigned short* p, int o0, int o1) {
  ABPack v;
  v.q[0] = *(const uint4*)(p + o0);
  v.q[1] = *(const uint4*)(p + o1);
  return v.v;
}

// ---------------------------------------------------------------------------
// Prep 1: build W^T in bf16, shape [H_][KAUG]:
//   Wt[n][k] = (k < H_) ? w_res[k][n] : w_in[k-H_][n]
// ---------------------------------------------------------------------------
__global__ void __launch_bounds__(256) esn_prep_w(const float* __restrict__ w_res,
                                                  const float* __restrict__ w_in,
                                                  unsigned short* __restrict__ Wt) {
  int idx = blockIdx.x * 256 + threadIdx.x;
  if (idx >= H_ * KAUG) return;
  int n = idx / KAUG;
  int k = idx - n * KAUG;
  float v = (k < H_) ? w_res[(size_t)k * H_ + n]
                     : w_in[(size_t)(k - H_) * H_ + n];
  Wt[(size_t)n * KAUG + k] = f2bf(v);
}

// Prep 2: X f32 -> bf16, same flat [B_][T_][F_] layout
__global__ void __launch_bounds__(256) esn_prep_x(const float* __restrict__ X,
                                                  unsigned short* __restrict__ Xbf) {
  int idx = blockIdx.x * 256 + threadIdx.x;
  if (idx >= B_ * T_ * F_) return;
  Xbf[idx] = f2bf(X[idx]);
}

// ---------------------------------------------------------------------------
// Device-wide barrier: bar[0] = arrive count, bar[1] = release generation.
// ---------------------------------------------------------------------------
__device__ __forceinline__ void grid_sync(unsigned int* bar, unsigned int gen) {
  __threadfence();
  __syncthreads();
  if (threadIdx.x == 0) {
    unsigned int prev = __hip_atomic_fetch_add(&bar[0], 1u, __ATOMIC_ACQ_REL,
                                               __HIP_MEMORY_SCOPE_AGENT);
    if (prev + 1u == (unsigned int)NBLK * gen) {
      __hip_atomic_store(&bar[1], gen, __ATOMIC_RELEASE, __HIP_MEMORY_SCOPE_AGENT);
    }
    while (__hip_atomic_load(&bar[1], __ATOMIC_ACQUIRE, __HIP_MEMORY_SCOPE_AGENT) < gen) {
      __builtin_amdgcn_s_sleep(2);
    }
  }
  __syncthreads();
}

// ---------------------------------------------------------------------------
// Persistent scan kernel.
// Block bid owns output columns [bid*32, bid*32+32). 4 waves -> four 16x16
// C tiles. Carried f32 state lives in registers. W^T slab cached in LDS.
// K-reduction split over 4 independent accumulators to break the
// WMMA->WMMA RAW hazard chain (bf16 dependent-WMMA costs ~5 slots).
// ---------------------------------------------------------------------------
__global__ void __launch_bounds__(SCAN_THREADS, 1)
esn_scan(const unsigned short* __restrict__ Wt,   // [H_][KAUG] bf16 (W^T)
         const unsigned short* __restrict__ Xbf,  // [B_][T_][F_] bf16
         unsigned short* __restrict__ Sbuf,       // 2 x [B_][H_] bf16
         float* __restrict__ out,                 // [B_][T_][H_] f32
         unsigned int* __restrict__ bar) {
  __shared__ __align__(16) unsigned short WtLds[32 * LDSPITCH]; // 139,776 B

  const int lane  = threadIdx.x & 31;
  const int wave  = threadIdx.x >> 5;       // 0..3
  const int laneM = lane & 15;              // A row / B,N column within tile
  const int half  = lane >> 4;              // lane-half: K group / M+8 select
  const int m0    = (wave >> 1) * 16;       // row tile base (batch rows)
  const int ncol0 = blockIdx.x * 32;
  const int ncol  = ncol0 + (wave & 1) * 16 + laneM; // this lane's N column

  // ---- stage this block's 32 W^T rows (all KAUG columns) into LDS ----
  for (int idx = threadIdx.x; idx < 32 * (KAUG / 8); idx += SCAN_THREADS) {
    int r = idx / (KAUG / 8);
    int c = idx - r * (KAUG / 8);           // uint4 index within row
    *(uint4*)&WtLds[r * LDSPITCH + c * 8] =
        *(const uint4*)(Wt + (size_t)(ncol0 + r) * KAUG + c * 8);
  }
  __syncthreads();

  unsigned short* S0 = Sbuf;
  unsigned short* S1 = Sbuf + (size_t)B_ * H_;

  // B operand source: one (padded) LDS row per lane, lane = N column
  const unsigned short* brow = WtLds + (size_t)((wave & 1) * 16 + laneM) * LDSPITCH;
  const unsigned short* xrow0 = Xbf + (size_t)(m0 + laneM) * T_ * F_;

  // per-operand sub-offsets (ISA layouts)
  const int a0 = half * 8;                  // A: K {0..7 | 8..15}
  const int a1 = 16 + half * 8;             // A: K {16..23 | 24..31}
  const int b0 = half * 8;                  // B: K {0..7 | 8..15}
  const int b1 = 16 + half * 8;             // B: K {16..23 | 24..31}

  v8f sreg = {0.f, 0.f, 0.f, 0.f, 0.f, 0.f, 0.f, 0.f};   // carried f32 state

  for (int t = 0; t < T_; ++t) {
    const unsigned short* Sc = (t & 1) ? S1 : S0;
    unsigned short*       Sn = (t & 1) ? S0 : S1;

    v8f c0 = {0.f, 0.f, 0.f, 0.f, 0.f, 0.f, 0.f, 0.f};
    v8f c1 = c0, c2 = c0, c3 = c0;
    const unsigned short* arow = Sc + (size_t)(m0 + laneM) * H_;

    // ---- recurrent part: K = 0..2047, 4-way independent accumulation ----
#pragma unroll 2
    for (int kc = 0; kc < H_ / 32; kc += 4) {
      const int k0 = kc * 32;
      c0 = __builtin_amdgcn_wmma_f32_16x16x32_bf16(
          false, ldpack(arow, k0 + a0, k0 + a1),
          false, ldpack(brow, k0 + b0, k0 + b1), (short)0, c0, false, false);
      c1 = __builtin_amdgcn_wmma_f32_16x16x32_bf16(
          false, ldpack(arow, k0 + 32 + a0, k0 + 32 + a1),
          false, ldpack(brow, k0 + 32 + b0, k0 + 32 + b1), (short)0, c1, false, false);
      c2 = __builtin_amdgcn_wmma_f32_16x16x32_bf16(
          false, ldpack(arow, k0 + 64 + a0, k0 + 64 + a1),
          false, ldpack(brow, k0 + 64 + b0, k0 + 64 + b1), (short)0, c2, false, false);
      c3 = __builtin_amdgcn_wmma_f32_16x16x32_bf16(
          false, ldpack(arow, k0 + 96 + a0, k0 + 96 + a1),
          false, ldpack(brow, k0 + 96 + b0, k0 + 96 + b1), (short)0, c3, false, false);
    }

    // ---- input-projection part: K = 2048..2175 over X[:,t,:] (4 chunks) ----
    const unsigned short* xrow = xrow0 + (size_t)t * F_;
    c0 = __builtin_amdgcn_wmma_f32_16x16x32_bf16(
        false, ldpack(xrow, a0, a1),
        false, ldpack(brow, H_ + b0, H_ + b1), (short)0, c0, false, false);
    c1 = __builtin_amdgcn_wmma_f32_16x16x32_bf16(
        false, ldpack(xrow, 32 + a0, 32 + a1),
        false, ldpack(brow, H_ + 32 + b0, H_ + 32 + b1), (short)0, c1, false, false);
    c2 = __builtin_amdgcn_wmma_f32_16x16x32_bf16(
        false, ldpack(xrow, 64 + a0, 64 + a1),
        false, ldpack(brow, H_ + 64 + b0, H_ + 64 + b1), (short)0, c2, false, false);
    c3 = __builtin_amdgcn_wmma_f32_16x16x32_bf16(
        false, ldpack(xrow, 96 + a0, 96 + a1),
        false, ldpack(brow, H_ + 96 + b0, H_ + 96 + b1), (short)0, c3, false, false);

    v8f acc = (c0 + c1) + (c2 + c3);

    // ---- epilogue: branch-free tanh + leaky blend, write out + next state ----
#pragma unroll
    for (int r = 0; r < 8; ++r) {
      const int m = m0 + half * 8 + r;            // C/D layout: VGPR r <-> row
      float th = fast_tanh(acc[r]);
      float ns = 0.9f * sreg[r] + 0.1f * th;
      sreg[r] = ns;
      // streaming 256MB output: non-temporal, keep L2 for state/X
      __builtin_nontemporal_store(ns, &out[((size_t)m * T_ + t) * H_ + ncol]);
      Sn[(size_t)m * H_ + ncol] = f2bf(ns);
    }

    if (t != T_ - 1) grid_sync(bar, (unsigned int)(t + 1));
  }
}

// ---------------------------------------------------------------------------
// Workspace layout (bytes):
//   [0)                 W^T bf16 : H_*KAUG*2           = 8,912,896
//   [8,912,896)         X   bf16 : B_*T_*F_*2          = 8,388,608
//   [17,301,504)        S ping-pong bf16 : 2*B_*H_*2   =   262,144
//   [17,563,648)        barrier counters (zeroed each launch)
// ---------------------------------------------------------------------------
extern "C" void kernel_launch(void* const* d_in, const int* in_sizes, int n_in,
                              void* d_out, int out_size, void* d_ws, size_t ws_size,
                              hipStream_t stream) {
  const float* X     = (const float*)d_in[0];  // [32,1024,128]
  const float* w_in  = (const float*)d_in[1];  // [128,2048]
  const float* w_res = (const float*)d_in[2];  // [2048,2048]
  float* out = (float*)d_out;                  // [32,1024,2048]

  char* ws = (char*)d_ws;
  unsigned short* Wt   = (unsigned short*)(ws);
  unsigned short* Xbf  = (unsigned short*)(ws + 8912896);
  unsigned short* Sbuf = (unsigned short*)(ws + 8912896 + 8388608);
  unsigned int*   bar  = (unsigned int*)  (ws + 8912896 + 8388608 + 262144);

  // zero state buffers + barrier counters (stream-ordered, capture-safe)
  hipMemsetAsync(Sbuf, 0, 262144 + 256, stream);

  esn_prep_w<<<(H_ * KAUG + 255) / 256, 256, 0, stream>>>(w_res, w_in, Wt);
  esn_prep_x<<<(B_ * T_ * F_ + 255) / 256, 256, 0, stream>>>(X, Xbf);
  esn_scan<<<NBLK, SCAN_THREADS, 0, stream>>>(Wt, Xbf, Sbuf, out, bar);
}
